// Transformer_41145786695700
// MI455X (gfx1250) — matrix operand order repaired and checked by reading, not compiled, not run on hardware
//
#include <hip/hip_runtime.h>
#include <hip/hip_bf16.h>

// Problem constants (match reference)
#define Bsz   64
#define Tsz   351
#define Dsz   512
#define Hsz   8
#define DKsz  64
#define DFFsz 1024
#define Lsz   3
#define TPAD  352   // 22*16, padded T for WMMA tiling

typedef __bf16 v16bf __attribute__((ext_vector_type(16)));
typedef float  v8f   __attribute__((ext_vector_type(8)));

struct __align__(16) U128 { unsigned int a, b, c, d; };
union Frag { U128 u[2]; unsigned short s[16]; v16bf v; };

// CDNA5 async global->LDS copy (ASYNCcnt-tracked), via inline asm:
// the clang builtin exists but is declared with LangAS::cuda_device pointers that
// cannot be formed from generic HIP pointers, so we emit the instruction directly.
__device__ __forceinline__ void cp16_g2l(const unsigned short* g, unsigned short* l) {
#if defined(__AMDGCN__)
  unsigned lds_off = (unsigned)(unsigned long long)l;   // flat LDS ptr: low 32 bits = LDS addr
  unsigned long long ga = (unsigned long long)g;
  asm volatile("global_load_async_to_lds_b128 %0, %1, off"
               :
               : "v"(lds_off), "v"(ga)
               : "memory");
#else
  *(U128*)l = *(const U128*)g;
#endif
}

__device__ __forceinline__ void wait_async_lds() {
#if defined(__AMDGCN__)
  asm volatile("s_wait_asynccnt 0" ::: "memory");
#endif
}

__device__ __forceinline__ unsigned short f2bf(float f) {
  unsigned int u = __float_as_uint(f);
  unsigned int r = (u + 0x7FFFu + ((u >> 16) & 1u)) >> 16;  // RNE
  return (unsigned short)r;
}

// ---------------- WMMA tile compute: wave computes 32x64 via 2x4 16x16 tiles ------------
// As: [128][32] bf16, Bs: [128][32] bf16 (both K-contiguous rows)
__device__ __forceinline__ void wmma_tile(const unsigned short* As, const unsigned short* Bs,
                                          v8f acc[2][4], int wm, int wn, int half, int l16) {
  Frag af[2], bfr[4];
#pragma unroll
  for (int mi = 0; mi < 2; ++mi) {
    int rl = wm * 32 + mi * 16 + l16;                         // A row (M) for this lane
    af[mi].u[0] = *(const U128*)&As[rl * 32 + half * 8];      // K = half*8 .. +7
    af[mi].u[1] = *(const U128*)&As[rl * 32 + 16 + half * 8]; // K = 16+half*8 .. +7
  }
#pragma unroll
  for (int ni = 0; ni < 4; ++ni) {
    int nl = wn * 64 + ni * 16 + l16;                          // B column (N) for this lane
    bfr[ni].u[0] = *(const U128*)&Bs[nl * 32 + half * 8];      // K = half*8 .. +7
    bfr[ni].u[1] = *(const U128*)&Bs[nl * 32 + 16 + half * 8]; // K = 16+half*8 .. +7
  }
#pragma unroll
  for (int mi = 0; mi < 2; ++mi)
#pragma unroll
    for (int ni = 0; ni < 4; ++ni)
      acc[mi][ni] = __builtin_amdgcn_wmma_f32_16x16x32_bf16(
          false, af[mi].v, false, bfr[ni].v, (short)0, acc[mi][ni], false, false);
}

// ---------------- Tile loaders: global bf16 -> LDS [128 rows][32 k] ---------------------
// Used for A tiles and for B stored [N,K] row-major.
__device__ __forceinline__ void load_tile_rk(unsigned short* Ls, const unsigned short* Gb,
                                             int r0, int k0, int R, int K, int ld) {
  int row  = threadIdx.x >> 1;       // 0..127
  int seg  = threadIdx.x & 1;        // 0..1 (16 elements each)
  int grow = r0 + row;
  int gk   = k0 + seg * 16;
  unsigned short* dst = &Ls[row * 32 + seg * 16];
  if (grow < R && gk + 16 <= K) {
    const unsigned short* srcp = Gb + (long long)grow * ld + gk;
    cp16_g2l(srcp, dst);
    cp16_g2l(srcp + 8, dst + 8);
    if (gk + 48 <= K) __builtin_prefetch(srcp + 32, 0, 0);   // next K-tile -> global_prefetch
  } else {
#pragma unroll
    for (int j = 0; j < 16; ++j)
      dst[j] = (grow < R && (gk + j) < K) ? Gb[(long long)grow * ld + gk + j] : (unsigned short)0;
  }
}

// B stored [K,N] row-major (e.g. V matrix for attn@V) -> transpose-gather into [N][K] LDS
__device__ __forceinline__ void load_tileB_k(unsigned short* Bs, const unsigned short* Bb,
                                             int n0, int k0, int N, int K, int ldb) {
  int n  = threadIdx.x & 127;
  int kg = (threadIdx.x >> 7) * 16;   // 0 or 16
#pragma unroll
  for (int j = 0; j < 16; ++j) {
    int kk = kg + j;
    Bs[n * 32 + kk] = ((n0 + n) < N && (k0 + kk) < K)
                          ? Bb[(long long)(k0 + kk) * ldb + (n0 + n)]
                          : (unsigned short)0;
  }
}

// ---------------- Generic batched WMMA GEMM (double-buffered LDS) -----------------------
// C[m,n] (+= bias[n]) = sum_k A[m,k]*B(.,.)  ; batched: z -> (b = z/Hdiv, h = z%Hdiv)
// EPI: 0 = f32 store, 1 = bf16 store, 2 = relu->bf16, 3 = f32 store of (acc + bias + res)
template <bool BKMAJ, int EPI>
__global__ __launch_bounds__(256) void gemm_wmma(
    int M, int N, int K, int Hdiv,
    const unsigned short* __restrict__ A, int lda, long long sAb, long long sAh,
    const unsigned short* __restrict__ Bm, int ldb, long long sBb, long long sBh,
    void* C, int ldc, long long sCb, long long sCh,
    const float* __restrict__ bias, const float* res) {
  __shared__ __align__(16) unsigned short As[2][128 * 32];
  __shared__ __align__(16) unsigned short Bs[2][128 * 32];
  int z  = blockIdx.z;
  int bz = z / Hdiv;
  int hz = z - bz * Hdiv;
  const unsigned short* Ab = A + (long long)bz * sAb + (long long)hz * sAh;
  const unsigned short* Bb = Bm + (long long)bz * sBb + (long long)hz * sBh;
  long long offC = (long long)bz * sCb + (long long)hz * sCh;
  int m0 = blockIdx.y * 128, n0 = blockIdx.x * 128;
  int lane = threadIdx.x & 31, wave = threadIdx.x >> 5;
  int wm = wave >> 1, wn = wave & 1, half = lane >> 4, l16 = lane & 15;

  v8f zero = {0.f, 0.f, 0.f, 0.f, 0.f, 0.f, 0.f, 0.f};
  v8f acc[2][4];
#pragma unroll
  for (int mi = 0; mi < 2; ++mi)
#pragma unroll
    for (int ni = 0; ni < 4; ++ni) acc[mi][ni] = zero;

  int nk = (K + 31) >> 5;
  // preload tile 0 into buffer 0
  load_tile_rk(As[0], Ab, m0, 0, M, K, lda);
  if (BKMAJ) load_tileB_k(Bs[0], Bb, n0, 0, N, K, ldb);
  else       load_tile_rk(Bs[0], Bb, n0, 0, N, K, ldb);

  for (int kt = 0; kt < nk; ++kt) {
    wait_async_lds();
    __syncthreads();
    int cur = kt & 1;
    if (kt + 1 < nk) {
      int k1 = (kt + 1) << 5;
      load_tile_rk(As[cur ^ 1], Ab, m0, k1, M, K, lda);
      if (BKMAJ) load_tileB_k(Bs[cur ^ 1], Bb, n0, k1, N, K, ldb);
      else       load_tile_rk(Bs[cur ^ 1], Bb, n0, k1, N, K, ldb);
    }
    wmma_tile(As[cur], Bs[cur], acc, wm, wn, half, l16);
  }

#pragma unroll
  for (int mi = 0; mi < 2; ++mi)
#pragma unroll
    for (int ni = 0; ni < 4; ++ni)
#pragma unroll
      for (int r = 0; r < 8; ++r) {
        int grow = m0 + wm * 32 + mi * 16 + half * 8 + r;
        int gcol = n0 + wn * 64 + ni * 16 + l16;
        if (grow < M && gcol < N) {
          float v = acc[mi][ni][r];
          if (bias) v += bias[gcol];
          long long ci = offC + (long long)grow * ldc + gcol;
          if (EPI == 0)      ((float*)C)[ci] = v;
          else if (EPI == 1) ((unsigned short*)C)[ci] = f2bf(v);
          else if (EPI == 2) ((unsigned short*)C)[ci] = f2bf(v > 0.f ? v : 0.f);
          else               ((float*)C)[ci] = v + res[ci];
        }
      }
}

// ---------------- Conv1d k=3 pad=1 as 3 accumulated WMMA GEMMs --------------------------
// out[b,t,d] = res[b,t,d] + b2[d] + sum_r sum_f W2r[r][d][f] * h[b,t+r-1,f]
__device__ __forceinline__ void conv_loadA(unsigned short* Ls, const unsigned short* Hm,
                                           int m0, int r, int k0, int M, int Tlen) {
  const int K = DFFsz;
  int row  = threadIdx.x >> 1;
  int seg  = threadIdx.x & 1;
  int grow = m0 + row;
  unsigned short* dst = &Ls[row * 32 + seg * 16];
  int bb = grow / Tlen;
  int tt = grow - bb * Tlen;
  int ts = tt + r - 1;
  if (grow < M && ts >= 0 && ts < Tlen) {
    const unsigned short* srcp = Hm + (long long)(bb * Tlen + ts) * K + k0 + seg * 16;
    cp16_g2l(srcp, dst);
    cp16_g2l(srcp + 8, dst + 8);
  } else {
#pragma unroll
    for (int j = 0; j < 16; ++j) dst[j] = 0;
  }
}

__global__ __launch_bounds__(256) void conv3_wmma(
    int M, int Tlen,
    const unsigned short* __restrict__ Hm,    // [M, DFF] bf16
    const unsigned short* __restrict__ W,     // [3][D][DFF] bf16
    const float* __restrict__ bias, const float* res, float* out) {
  const int N = Dsz, K = DFFsz;
  __shared__ __align__(16) unsigned short As[2][128 * 32];
  __shared__ __align__(16) unsigned short Bs[2][128 * 32];
  int m0 = blockIdx.y * 128, n0 = blockIdx.x * 128;
  int lane = threadIdx.x & 31, wave = threadIdx.x >> 5;
  int wm = wave >> 1, wn = wave & 1, half = lane >> 4, l16 = lane & 15;

  v8f zero = {0.f, 0.f, 0.f, 0.f, 0.f, 0.f, 0.f, 0.f};
  v8f acc[2][4];
#pragma unroll
  for (int mi = 0; mi < 2; ++mi)
#pragma unroll
    for (int ni = 0; ni < 4; ++ni) acc[mi][ni] = zero;

  const int nk = K >> 5;           // 32
  const int total = 3 * nk;        // 96 iterations over (r, k0)
  // preload iteration 0
  conv_loadA(As[0], Hm, m0, 0, 0, M, Tlen);
  load_tile_rk(Bs[0], W, n0, 0, N, K, K);

  for (int it = 0; it < total; ++it) {
    wait_async_lds();
    __syncthreads();
    int cur = it & 1;
    if (it + 1 < total) {
      int r1 = (it + 1) / nk;
      int k1 = ((it + 1) - r1 * nk) << 5;
      conv_loadA(As[cur ^ 1], Hm, m0, r1, k1, M, Tlen);
      load_tile_rk(Bs[cur ^ 1], W + (long long)r1 * N * K, n0, k1, N, K, K);
    }
    wmma_tile(As[cur], Bs[cur], acc, wm, wn, half, l16);
  }

#pragma unroll
  for (int mi = 0; mi < 2; ++mi)
#pragma unroll
    for (int ni = 0; ni < 4; ++ni)
#pragma unroll
      for (int r = 0; r < 8; ++r) {
        int grow = m0 + wm * 32 + mi * 16 + half * 8 + r;
        int gcol = n0 + wn * 64 + ni * 16 + l16;
        if (grow < M && gcol < N) {
          long long ci = (long long)grow * Dsz + gcol;
          out[ci] = res[ci] + acc[mi][ni][r] + bias[gcol];
        }
      }
}

// ---------------- Fused (x += add) + LayerNorm (unbiased std, /(std+eps)) ---------------
__global__ __launch_bounds__(128) void add_ln(
    const float* __restrict__ src, const float* __restrict__ add, int addMod,
    float* xw_out, const float* __restrict__ gamma, const float* __restrict__ beta,
    unsigned short* out_bf, float* out_f32) {
  int row = blockIdx.x;
  int t = threadIdx.x;
  __shared__ float red[128];
  const float* s = src + (long long)row * Dsz;
  int tmod = row % addMod;
  float v[4];
#pragma unroll
  for (int i = 0; i < 4; ++i) {
    int j = t + i * 128;
    float x = s[j];
    if (add) x += add[(long long)tmod * Dsz + j];
    v[i] = x;
    if (xw_out) xw_out[(long long)row * Dsz + j] = x;
  }
  float ls = v[0] + v[1] + v[2] + v[3];
  red[t] = ls; __syncthreads();
  for (int st = 64; st > 0; st >>= 1) { if (t < st) red[t] += red[t + st]; __syncthreads(); }
  float mean = red[0] * (1.0f / Dsz);
  __syncthreads();
  float lq = 0.f;
#pragma unroll
  for (int i = 0; i < 4; ++i) { float d = v[i] - mean; lq += d * d; }
  red[t] = lq; __syncthreads();
  for (int st = 64; st > 0; st >>= 1) { if (t < st) red[t] += red[t + st]; __syncthreads(); }
  float denom = sqrtf(red[0] * (1.0f / (Dsz - 1))) + 1e-6f;
#pragma unroll
  for (int i = 0; i < 4; ++i) {
    int j = t + i * 128;
    float g = gamma[j] * (v[i] - mean) / denom + beta[j];
    if (out_bf)  out_bf[(long long)row * Dsz + j] = f2bf(g);
    if (out_f32) out_f32[(long long)row * Dsz + j] = g;
  }
}

// ---------------- rel_shift + softmax fused ---------------------------------------------
__global__ __launch_bounds__(128) void softmax_shift(
    const float* __restrict__ S1, const float* __restrict__ S2,
    unsigned short* __restrict__ attn) {
  int q = blockIdx.x;                 // 0..T-1
  long long z = blockIdx.y;           // B*H
  const float* s1 = S1 + z * (long long)(TPAD * TPAD) + (long long)q * TPAD;
  const float* s2 = S2 + z * (long long)(TPAD * TPAD);
  __shared__ float red[128];
  int t = threadIdx.x;
  float vals[3];
#pragma unroll
  for (int i = 0; i < 3; ++i) {
    int k = t + i * 128;
    float v = -3.0e38f;
    if (k < Tsz) {
      int g  = q * Tsz + k + Tsz;     // rel_shift index math (pad+reshape)
      int qp = g / (Tsz + 1);
      int j  = g - qp * (Tsz + 1);
      float pv = (j == 0) ? 0.f : s2[(long long)qp * TPAD + (j - 1)];
      v = (s1[k] + pv) * 0.044194173824159216f;   // 1/sqrt(512)
    }
    vals[i] = v;
  }
  float m = fmaxf(vals[0], fmaxf(vals[1], vals[2]));
  red[t] = m; __syncthreads();
  for (int st = 64; st > 0; st >>= 1) { if (t < st) red[t] = fmaxf(red[t], red[t + st]); __syncthreads(); }
  float mx = red[0]; __syncthreads();
  float e[3]; float ls = 0.f;
#pragma unroll
  for (int i = 0; i < 3; ++i) {
    e[i] = (vals[i] > -1.0e38f) ? expf(vals[i] - mx) : 0.f;
    ls += e[i];
  }
  red[t] = ls; __syncthreads();
  for (int st = 64; st > 0; st >>= 1) { if (t < st) red[t] += red[t + st]; __syncthreads(); }
  float inv = 1.f / red[0];
  unsigned short* arow = attn + z * (long long)(TPAD * TPAD) + (long long)q * TPAD;
#pragma unroll
  for (int i = 0; i < 3; ++i) {
    int k = t + i * 128;
    if (k < Tsz) arow[k] = f2bf(e[i] * inv);
  }
}

// ---------------- small elementwise kernels ---------------------------------------------
__global__ void cvt_bf16(const float* __restrict__ in, unsigned short* __restrict__ out, int n) {
  int i = blockIdx.x * blockDim.x + threadIdx.x;
  if (i < n) out[i] = f2bf(in[i]);
}

__global__ void build_quqv(const float* __restrict__ q, const float* __restrict__ u,
                           const float* __restrict__ v, unsigned short* __restrict__ qu,
                           unsigned short* __restrict__ qv, int n) {
  int i = blockIdx.x * blockDim.x + threadIdx.x;
  if (i < n) {
    int d = i & (Dsz - 1);            // u_bias[H,DK] flattens to [D]
    float qx = q[i];
    qu[i] = f2bf(qx + u[d]);
    qv[i] = f2bf(qx + v[d]);
  }
}

// out[((l*3+r)*D + d)*DFF + f] = W2[l,d,f,r]
__global__ void repack_w2(const float* __restrict__ w2, unsigned short* __restrict__ out, int n) {
  int i = blockIdx.x * blockDim.x + threadIdx.x;
  if (i < n) {
    int f = i & (DFFsz - 1);
    int rest = i >> 10;
    int d = rest & (Dsz - 1);
    rest >>= 9;
    int r = rest % 3;
    int l = rest / 3;
    out[i] = f2bf(w2[(((long long)l * Dsz + d) * DFFsz + f) * 3 + r]);
  }
}

// ---------------- launcher ---------------------------------------------------------------
extern "C" void kernel_launch(void* const* d_in, const int* in_sizes, int n_in,
                              void* d_out, int out_size, void* d_ws, size_t ws_size,
                              hipStream_t stream) {
  (void)in_sizes; (void)n_in; (void)out_size; (void)ws_size;
  const float* x_in = (const float*)d_in[0];
  const float* mask = (const float*)d_in[1];
  const float* pos  = (const float*)d_in[2];
  const float* Wq = (const float*)d_in[3];  const float* bq = (const float*)d_in[4];
  const float* Wk = (const float*)d_in[5];  const float* bk = (const float*)d_in[6];
  const float* Wv = (const float*)d_in[7];  const float* bv = (const float*)d_in[8];
  const float* Wp = (const float*)d_in[9];
  const float* ub = (const float*)d_in[10]; const float* vbias = (const float*)d_in[11];
  const float* Wo = (const float*)d_in[12]; const float* bo = (const float*)d_in[13];
  const float* l1a = (const float*)d_in[14]; const float* l1b = (const float*)d_in[15];
  const float* l2a = (const float*)d_in[16]; const float* l2b = (const float*)d_in[17];
  const float* W1 = (const float*)d_in[18]; const float* b1 = (const float*)d_in[19];
  const float* W2 = (const float*)d_in[20]; const float* b2 = (const float*)d_in[21];
  const float* fna = (const float*)d_in[22]; const float* fnb = (const float*)d_in[23];

  const int  BT  = Bsz * Tsz;                      // 22464
  const long long BTD = (long long)BT * Dsz;       // 11,501,568
  const long long SS  = (long long)Bsz * Hsz * TPAD * TPAD;  // score elems
  const int  DD  = Dsz * Dsz;
  const long long sBT = (long long)Tsz * Dsz;      // per-batch stride of [B,T,D]
  const long long sSS = (long long)TPAD * TPAD;

  char* wsb = (char*)d_ws;
  size_t off = 0;
  auto take = [&](size_t bytes) -> char* {
    char* p = wsb + off;
    off = (off + bytes + 255) & ~(size_t)255;
    return p;
  };
  float* xw   = (float*)take(BTD * 4);
  float* qf   = (float*)take(BTD * 4);
  float* S1   = (float*)take(SS * 4);
  float* S2   = (float*)take(SS * 4);
  unsigned short* attn = (unsigned short*)take(SS * 2);
  unsigned short* xn = (unsigned short*)take(BTD * 2);
  unsigned short* pe = (unsigned short*)take(BTD * 2);
  unsigned short* qu = (unsigned short*)take(BTD * 2);
  unsigned short* qv = (unsigned short*)take(BTD * 2);
  unsigned short* kb = (unsigned short*)take(BTD * 2);
  unsigned short* pb = (unsigned short*)take(BTD * 2);
  unsigned short* vbuf = (unsigned short*)take(BTD * 2);
  unsigned short* ctx  = (unsigned short*)take(BTD * 2);
  unsigned short* hb   = (unsigned short*)take((long long)BT * DFFsz * 2);
  unsigned short* wq_bf = (unsigned short*)take((long long)Lsz * DD * 2);
  unsigned short* wk_bf = (unsigned short*)take((long long)Lsz * DD * 2);
  unsigned short* wv_bf = (unsigned short*)take((long long)Lsz * DD * 2);
  unsigned short* wp_bf = (unsigned short*)take((long long)Lsz * DD * 2);
  unsigned short* wo_bf = (unsigned short*)take((long long)Lsz * DD * 2);
  unsigned short* w1_bf = (unsigned short*)take((long long)Lsz * DFFsz * Dsz * 2);
  unsigned short* w2r   = (unsigned short*)take((long long)Lsz * 3 * Dsz * DFFsz * 2);

  auto cd = [](long long a, long long b) -> unsigned { return (unsigned)((a + b - 1) / b); };
  dim3 blk(256);

  // --- conversions (re-done each launch; deterministic) ---
  {
    int nw = Lsz * DD;
    cvt_bf16<<<cd(nw, 256), 256, 0, stream>>>(Wq, wq_bf, nw);
    cvt_bf16<<<cd(nw, 256), 256, 0, stream>>>(Wk, wk_bf, nw);
    cvt_bf16<<<cd(nw, 256), 256, 0, stream>>>(Wv, wv_bf, nw);
    cvt_bf16<<<cd(nw, 256), 256, 0, stream>>>(Wp, wp_bf, nw);
    cvt_bf16<<<cd(nw, 256), 256, 0, stream>>>(Wo, wo_bf, nw);
    int n1 = Lsz * DFFsz * Dsz;
    cvt_bf16<<<cd(n1, 256), 256, 0, stream>>>(W1, w1_bf, n1);
    cvt_bf16<<<cd(BTD, 256), 256, 0, stream>>>(mask, pe, (int)BTD);
    int n2 = Lsz * 3 * Dsz * DFFsz;
    repack_w2<<<cd(n2, 256), 256, 0, stream>>>(W2, w2r, n2);
  }

  for (int i = 0; i < Lsz; ++i) {
    // 1) x += pos_emb[i]; xn = LN1(x)  (bf16)
    add_ln<<<BT, 128, 0, stream>>>((i == 0) ? x_in : xw, pos + (long long)i * Tsz * Dsz, Tsz,
                                   xw, l1a + i * Dsz, l1b + i * Dsz, xn, nullptr);

    // 2) projections: q(f32), k/v/p (bf16)
    gemm_wmma<false, 0><<<dim3(4, 176, 1), blk, 0, stream>>>(
        BT, Dsz, Dsz, 1, xn, Dsz, 0, 0, wq_bf + (long long)i * DD, Dsz, 0, 0,
        qf, Dsz, 0, 0, bq + i * Dsz, nullptr);
    gemm_wmma<false, 1><<<dim3(4, 176, 1), blk, 0, stream>>>(
        BT, Dsz, Dsz, 1, xn, Dsz, 0, 0, wk_bf + (long long)i * DD, Dsz, 0, 0,
        kb, Dsz, 0, 0, bk + i * Dsz, nullptr);
    gemm_wmma<false, 1><<<dim3(4, 176, 1), blk, 0, stream>>>(
        BT, Dsz, Dsz, 1, xn, Dsz, 0, 0, wv_bf + (long long)i * DD, Dsz, 0, 0,
        vbuf, Dsz, 0, 0, bv + i * Dsz, nullptr);
    gemm_wmma<false, 1><<<dim3(4, 176, 1), blk, 0, stream>>>(
        BT, Dsz, Dsz, 1, pe, Dsz, 0, 0, wp_bf + (long long)i * DD, Dsz, 0, 0,
        pb, Dsz, 0, 0, nullptr, nullptr);

    // 3) qu = bf16(q + u), qv = bf16(q + v)
    build_quqv<<<cd(BTD, 256), 256, 0, stream>>>(qf, ub + i * Dsz, vbias + i * Dsz, qu, qv, (int)BTD);

    // 4) scores: S1 = (q+u) @ k^T, S2 = (q+v) @ p^T  (batched over B*H)
    gemm_wmma<false, 0><<<dim3(3, 3, Bsz * Hsz), blk, 0, stream>>>(
        Tsz, Tsz, DKsz, Hsz, qu, Dsz, sBT, DKsz, kb, Dsz, sBT, DKsz,
        S1, TPAD, (long long)Hsz * sSS, sSS, nullptr, nullptr);
    gemm_wmma<false, 0><<<dim3(3, 3, Bsz * Hsz), blk, 0, stream>>>(
        Tsz, Tsz, DKsz, Hsz, qv, Dsz, sBT, DKsz, pb, Dsz, sBT, DKsz,
        S2, TPAD, (long long)Hsz * sSS, sSS, nullptr, nullptr);

    // 5) attn = softmax((S1 + rel_shift(S2)) / sqrt(D))
    softmax_shift<<<dim3(Tsz, Bsz * Hsz), 128, 0, stream>>>(S1, S2, attn);

    // 6) ctx = attn @ v   (B is K-major here)
    gemm_wmma<true, 1><<<dim3(1, 3, Bsz * Hsz), blk, 0, stream>>>(
        Tsz, DKsz, Tsz, Hsz, attn, TPAD, (long long)Hsz * sSS, sSS,
        vbuf, Dsz, sBT, DKsz, ctx, Dsz, sBT, DKsz, nullptr, nullptr);

    // 7) x += ctx @ Wo^T + bo
    gemm_wmma<false, 3><<<dim3(4, 176, 1), blk, 0, stream>>>(
        BT, Dsz, Dsz, 1, ctx, Dsz, 0, 0, wo_bf + (long long)i * DD, Dsz, 0, 0,
        xw, Dsz, 0, 0, bo + i * Dsz, xw);

    // 8) xn = LN2(x)
    add_ln<<<BT, 128, 0, stream>>>(xw, nullptr, 1, nullptr, l2a + i * Dsz, l2b + i * Dsz, xn, nullptr);

    // 9) h = relu(xn @ W1^T + b1) (bf16)
    gemm_wmma<false, 2><<<dim3(8, 176, 1), blk, 0, stream>>>(
        BT, DFFsz, Dsz, 1, xn, Dsz, 0, 0, w1_bf + (long long)i * DFFsz * Dsz, Dsz, 0, 0,
        hb, DFFsz, 0, 0, b1 + i * DFFsz, nullptr);

    // 10) x += conv1d_k3(h) + b2
    conv3_wmma<<<dim3(4, 176, 1), blk, 0, stream>>>(
        BT, Tsz, hb, w2r + (long long)i * 3 * Dsz * DFFsz, b2 + i * Dsz, xw, xw);
  }

  // final layernorm -> d_out (f32)
  add_ln<<<BT, 128, 0, stream>>>(xw, nullptr, 1, nullptr, fna, fnb, nullptr, (float*)d_out);
}
